// DriveNeuralNetworkWithMemory_73564199845966
// MI455X (gfx1250) — matrix operand order repaired and checked by reading, not compiled, not run on hardware
//
#include <hip/hip_runtime.h>
#include <hip/hip_bf16.h>

typedef __attribute__((ext_vector_type(16))) _Float16 v16h;
typedef __attribute__((ext_vector_type(8)))  _Float16 v8h;
typedef __attribute__((ext_vector_type(8)))  float    v8f;

#define T_SEQ 256
#define F_IN  68
#define HL    64     // LSTM hidden
#define GATES 256    // 4*HL
#define KTOT  96     // 32 (h1 padded) + 64 (h)

// Branch-free hardware-rate nonlinearities (v_exp_f32 + v_rcp_f32, no divides,
// no EXEC-mask branches) for the serial LSTM update path.
__device__ __forceinline__ float fast_sigmoid(float x) {
    return __builtin_amdgcn_rcpf(1.f + __expf(-x));
}
__device__ __forceinline__ float fast_tanh(float x) {
    return 1.f - 2.f * __builtin_amdgcn_rcpf(1.f + __expf(2.f * x));
}

// ---------------------------------------------------------------------------
// Kernel 1: pointwise feature MLPs -> h1[B*T][32] (f16, slots 25..31 zero)
// ---------------------------------------------------------------------------
__global__ __launch_bounds__(256) void k_features(
    const float* __restrict__ x,
    const float* __restrict__ W_acc1, const float* __restrict__ b_acc1,
    const float* __restrict__ W_sh,   const float* __restrict__ b_sh,
    const float* __restrict__ W_st,   const float* __restrict__ b_st,
    const float* __restrict__ W_acc2, const float* __restrict__ b_acc2,
    const float* __restrict__ W_s2,   const float* __restrict__ b_s2,
    _Float16* __restrict__ h1_out, int npts)
{
    // LDS weight stage: Wacc1[0,256) bacc1[256,272) Wsh[272,784) bsh[784,816)
    // Wst[816,1616) bst[1616,1632) Wacc2[1632,1888) bacc2[1888,1904)
    // Ws2[1904,2288) bs2[2288,2296)
    __shared__ float sW[2296];
    for (int i = threadIdx.x; i < 2296; i += 256) {
        float v;
        if      (i < 256)  v = W_acc1[i];
        else if (i < 272)  v = b_acc1[i - 256];
        else if (i < 784)  v = W_sh[i - 272];
        else if (i < 816)  v = b_sh[i - 784];
        else if (i < 1616) v = W_st[i - 816];
        else if (i < 1632) v = b_st[i - 1616];
        else if (i < 1888) v = W_acc2[i - 1632];
        else if (i < 1904) v = b_acc2[i - 1888];
        else if (i < 2288) v = W_s2[i - 1904];
        else               v = b_s2[i - 2288];
        sW[i] = v;
    }
    __syncthreads();

    int p = blockIdx.x * 256 + threadIdx.x;
    if (p >= npts) return;
    const float* xr = x + (size_t)p * F_IN;

    float acc_in[16], st_h[16], st_f[50];
    acc_in[0] = xr[1];            // ACC_IDX = [1, 3..17], ACC_NORM = 1
    st_h[0]   = 2.f * xr[2];      // STEER_IDX = [2, 18..32], STEER_NORM = 2
#pragma unroll
    for (int j = 1; j < 16; ++j) { acc_in[j] = xr[2 + j]; st_h[j] = 2.f * xr[17 + j]; }
    for (int j = 0; j < 50; ++j) st_f[j] = 2.f * xr[18 + j];   // STEER_IDX_FULL = [18..67]

    float acc1[16];
    for (int i = 0; i < 16; ++i) {
        float s = sW[256 + i];
        for (int k = 0; k < 16; ++k) s += sW[i * 16 + k] * acc_in[k];
        acc1[i] = fmaxf(s, 0.f);
    }
    float s1[48];
    for (int i = 0; i < 32; ++i) {
        float s = sW[784 + i];
        for (int k = 0; k < 16; ++k) s += sW[272 + i * 16 + k] * st_h[k];
        s1[i] = fmaxf(s, 0.f);
    }
    for (int i = 0; i < 16; ++i) {
        float s = sW[1616 + i];
        for (int k = 0; k < 50; ++k) s += sW[816 + i * 50 + k] * st_f[k];
        s1[32 + i] = fmaxf(s, 0.f);
    }
    float h1[32];
    h1[0] = 0.1f * xr[0];                       // VEL_NORM
    for (int i = 0; i < 16; ++i) {              // acc2
        float s = sW[1888 + i];
        for (int k = 0; k < 16; ++k) s += sW[1632 + i * 16 + k] * acc1[k];
        h1[1 + i] = fmaxf(s, 0.f);
    }
    for (int i = 0; i < 8; ++i) {               // s2
        float s = sW[2288 + i];
        for (int k = 0; k < 48; ++k) s += sW[1904 + i * 48 + k] * s1[k];
        h1[17 + i] = fmaxf(s, 0.f);
    }
    for (int i = 25; i < 32; ++i) h1[i] = 0.f;  // zero pad (WMMA K pad)

    _Float16* o = h1_out + (size_t)p * 32;
    for (int i = 0; i < 32; ++i) o[i] = (_Float16)h1[i];
}

// ---------------------------------------------------------------------------
// Kernel 2: LSTM scan with WMMA. One block = 16 batch rows x all T steps.
// 4 waves; wave w computes gate columns [64w, 64w+64).
// B (weight) fragments are loaded to registers ONCE and held across the scan.
// ---------------------------------------------------------------------------
__global__ __launch_bounds__(128) void k_lstm(
    const float* __restrict__ W_ih, const float* __restrict__ W_hh,
    const float* __restrict__ b_ih, const float* __restrict__ b_hh,
    const _Float16* __restrict__ h1_in,
    _Float16* __restrict__ hs_out)
{
    __shared__ __align__(32) _Float16 sB[GATES * KTOT];   // B matrix [N][K], 48 KB
    __shared__ float    sBias[GATES];                      // b_ih + b_hh
    __shared__ float    sGates[16 * GATES];                // raw gates, 16 KB
    __shared__ float    sC[16 * HL];                       // cell state
    __shared__ __align__(32) _Float16 sH[16 * HL];         // hidden (f16)

    const int tid = threadIdx.x;
    // Stage combined weights: per column n, K 0..24 -> W_ih, 25..31 -> 0, 32..95 -> W_hh
    for (int n = tid; n < GATES; n += 128) {
        for (int k = 0; k < 25; ++k)  sB[n * KTOT + k]      = (_Float16)W_ih[n * 25 + k];
        for (int k = 25; k < 32; ++k) sB[n * KTOT + k]      = (_Float16)0.f;
        for (int k = 0; k < 64; ++k)  sB[n * KTOT + 32 + k] = (_Float16)W_hh[n * 64 + k];
        sBias[n] = b_ih[n] + b_hh[n];
    }
    for (int i = tid; i < 16 * HL; i += 128) { sC[i] = 0.f; sH[i] = (_Float16)0.f; }
    __syncthreads();

    const int lane  = tid & 31;
    const int wv    = tid >> 5;     // wave id 0..3
    const int hf    = lane >> 4;    // K-half select (A/B fragment layout)
    const int rw    = lane & 15;    // row (A) / column-in-tile (B, C/D)
    const int bbase = blockIdx.x * 16;

    // ---- hoist loop-invariant B fragments + biases into registers
    v16h  Bf[12];
    float biasv[4];
#pragma unroll
    for (int nt = 0; nt < 4; ++nt) {
        const int nb = (wv * 4 + nt) * 16;
        const _Float16* bp = &sB[(size_t)(nb + rw) * KTOT];
        Bf[nt * 3 + 0] = *(const v16h*)(bp + hf * 16);        // K  0..31 (h1)
        Bf[nt * 3 + 1] = *(const v16h*)(bp + 32 + hf * 16);   // K 32..63 (h lo)
        Bf[nt * 3 + 2] = *(const v16h*)(bp + 64 + hf * 16);   // K 64..95 (h hi)
        biasv[nt] = sBias[nb + rw];
    }

    const _Float16* h1base = h1_in + (((size_t)(bbase + rw)) * T_SEQ) * 32;

    for (int t = 0; t < T_SEQ; ++t) {
        // ---- A fragments (contiguous K order chosen here, same as B: VGPR0-7 = K hf*16..+15)
        union { v16h v; v8h h[2]; } a0, a1, a2;
        const _Float16* h1p = h1base + (size_t)t * 32;
        a0.h[0] = *(const v8h*)(h1p + hf * 16);
        a0.h[1] = *(const v8h*)(h1p + 8 + hf * 16);
        if (t + 1 < T_SEQ)                                    // global_prefetch_b8
            __builtin_prefetch(h1p + 32, 0, 3);
        const _Float16* hp = &sH[rw * HL];
        a1.h[0] = *(const v8h*)(hp + hf * 16);
        a1.h[1] = *(const v8h*)(hp + 8 + hf * 16);
        a2.h[0] = *(const v8h*)(hp + 32 + hf * 16);
        a2.h[1] = *(const v8h*)(hp + 40 + hf * 16);

#pragma unroll
        for (int nt = 0; nt < 4; ++nt) {
            const int nb = (wv * 4 + nt) * 16;          // gate column base
            v8f acc;
#pragma unroll
            for (int r = 0; r < 8; ++r) acc[r] = biasv[nt];   // bias depends only on N
            acc = __builtin_amdgcn_wmma_f32_16x16x32_f16(false, a0.v, false, Bf[nt*3+0], (short)0, acc, false, false);
            acc = __builtin_amdgcn_wmma_f32_16x16x32_f16(false, a1.v, false, Bf[nt*3+1], (short)0, acc, false, false);
            acc = __builtin_amdgcn_wmma_f32_16x16x32_f16(false, a2.v, false, Bf[nt*3+2], (short)0, acc, false, false);
#pragma unroll
            for (int r = 0; r < 8; ++r)                  // C/D layout: M = r + 8*hf, N = rw
                sGates[(r + 8 * hf) * GATES + nb + rw] = acc[r];
        }
        __syncthreads();

        // ---- nonlinear update: 16 rows x 64 hidden = 1024 values / 128 threads
#pragma unroll
        for (int u = 0; u < 8; ++u) {
            int idx = tid * 8 + u; int row = idx >> 6; int j = idx & 63;
            float gi = sGates[row * GATES + j];
            float gf = sGates[row * GATES + 64 + j];
            float gg = sGates[row * GATES + 128 + j];
            float go = sGates[row * GATES + 192 + j];
            float c  = fast_sigmoid(gf) * sC[row * HL + j]
                     + fast_sigmoid(gi) * fast_tanh(gg);
            float h  = fast_sigmoid(go) * fast_tanh(c);
            sC[row * HL + j] = c;
            sH[row * HL + j] = (_Float16)h;
            hs_out[(((size_t)(bbase + row)) * T_SEQ + t) * HL + j] = (_Float16)h;
        }
        __syncthreads();
    }
}

// ---------------------------------------------------------------------------
// Kernel 3: output heads (l1 / l2 / final) per (b,t) point.
// ---------------------------------------------------------------------------
__global__ __launch_bounds__(256) void k_head(
    const float* __restrict__ W_l1, const float* __restrict__ b_l1,
    const float* __restrict__ W_l2, const float* __restrict__ b_l2,
    const float* __restrict__ W_f,  const float* __restrict__ b_f,
    const _Float16* __restrict__ h1_in, const _Float16* __restrict__ hs_in,
    float* __restrict__ out, int npts)
{
    // W_l1[0,800) b_l1[800,832) W_l2[832,2368) b_l2[2368,2384) W_f[2384,2624) b_f[2624,2630)
    __shared__ float sW[2630];
    for (int i = threadIdx.x; i < 2630; i += 256) {
        float v;
        if      (i < 800)  v = W_l1[i];
        else if (i < 832)  v = b_l1[i - 800];
        else if (i < 2368) v = W_l2[i - 832];
        else if (i < 2384) v = b_l2[i - 2368];
        else if (i < 2624) v = W_f[i - 2384];
        else               v = b_f[i - 2624];
        sW[i] = v;
    }
    __syncthreads();

    int p = blockIdx.x * 256 + threadIdx.x;
    if (p >= npts) return;
    const _Float16* h1p = h1_in + (size_t)p * 32;
    const _Float16* hp  = hs_in + (size_t)p * HL;

    float h1[25];
    for (int k = 0; k < 25; ++k) h1[k] = (float)h1p[k];
    float cat[96];
    for (int k = 0; k < 64; ++k) cat[k] = (float)hp[k];
    for (int i = 0; i < 32; ++i) {                       // l1 = relu(h1 @ W_l1ᵀ)
        float s = sW[800 + i];
        for (int k = 0; k < 25; ++k) s += sW[i * 25 + k] * h1[k];
        cat[64 + i] = fmaxf(s, 0.f);
    }
    float fin[40];
    for (int i = 0; i < 16; ++i) {                       // l2 = relu([lstm,l1] @ W_l2ᵀ)
        float s = sW[2368 + i];
        for (int k = 0; k < 96; ++k) s += sW[832 + i * 96 + k] * cat[k];
        fin[i] = fmaxf(s, 0.f);
    }
    for (int k = 0; k < 24; ++k) fin[16 + k] = h1[1 + k];  // acc2(16) + s2(8)
    float* op = out + (size_t)p * 6;
    for (int i = 0; i < 6; ++i) {
        float s = sW[2624 + i];
        for (int k = 0; k < 40; ++k) s += sW[2384 + i * 40 + k] * fin[k];
        op[i] = s;
    }
}

// ---------------------------------------------------------------------------
extern "C" void kernel_launch(void* const* d_in, const int* in_sizes, int n_in,
                              void* d_out, int out_size, void* d_ws, size_t ws_size,
                              hipStream_t stream)
{
    const float* x      = (const float*)d_in[0];
    const float* W_acc1 = (const float*)d_in[1];  const float* b_acc1 = (const float*)d_in[2];
    const float* W_sh   = (const float*)d_in[3];  const float* b_sh   = (const float*)d_in[4];
    const float* W_st   = (const float*)d_in[5];  const float* b_st   = (const float*)d_in[6];
    const float* W_acc2 = (const float*)d_in[7];  const float* b_acc2 = (const float*)d_in[8];
    const float* W_s2   = (const float*)d_in[9];  const float* b_s2   = (const float*)d_in[10];
    const float* W_ih   = (const float*)d_in[11]; const float* W_hh   = (const float*)d_in[12];
    const float* b_ih   = (const float*)d_in[13]; const float* b_hh   = (const float*)d_in[14];
    const float* W_l1   = (const float*)d_in[15]; const float* b_l1   = (const float*)d_in[16];
    const float* W_l2   = (const float*)d_in[17]; const float* b_l2   = (const float*)d_in[18];
    const float* W_f    = (const float*)d_in[19]; const float* b_f    = (const float*)d_in[20];

    const int npts = in_sizes[0] / F_IN;   // B*T
    const int B    = npts / T_SEQ;

    _Float16* h1_ws = (_Float16*)d_ws;                                   // npts*32 f16
    _Float16* hs_ws = (_Float16*)((char*)d_ws + (size_t)npts * 32 * 2);  // npts*64 f16

    k_features<<<(npts + 255) / 256, 256, 0, stream>>>(
        x, W_acc1, b_acc1, W_sh, b_sh, W_st, b_st, W_acc2, b_acc2, W_s2, b_s2,
        h1_ws, npts);

    k_lstm<<<B / 16, 128, 0, stream>>>(W_ih, W_hh, b_ih, b_hh, h1_ws, hs_ws);

    k_head<<<(npts + 255) / 256, 256, 0, stream>>>(
        W_l1, b_l1, W_l2, b_l2, W_f, b_f, h1_ws, hs_ws, (float*)d_out, npts);
}
